// BiGRU_1486058685008
// MI455X (gfx1250) — matrix operand order repaired and checked by reading, not compiled, not run on hardware
//
#include <hip/hip_runtime.h>
#include <hip/hip_bf16.h>

// ---------- types ----------
typedef __attribute__((ext_vector_type(16))) __bf16 v16bf;
typedef __attribute__((ext_vector_type(8)))  __bf16 v8bf;
typedef __attribute__((ext_vector_type(8)))  float  v8f;

// Problem constants
#define SEQ   1024
#define BATCH 16
#define DIN   768        // patch dim == 2H
#define HID   384
#define G3    1152       // 3*HID
#define MROWS (SEQ*BATCH)    // 16384
#define NSLICE 4             // WGPs per direction in the scan
#define JSL   (HID/NSLICE)   // 96 hidden units per slice
#define SCAN_THREADS 576     // 18 waves = 18 N-tiles per slice

// ---------------------------------------------------------------------------
// fp32 -> bf16 elementwise convert
// ---------------------------------------------------------------------------
__global__ void cvt_f32_bf16_kernel(const float* __restrict__ src,
                                    __bf16* __restrict__ dst, int n) {
    int i = blockIdx.x * blockDim.x + threadIdx.x;
    if (i < n) dst[i] = (__bf16)src[i];
}

__global__ void zero_i32_kernel(int* __restrict__ p, int n) {
    int i = blockIdx.x * blockDim.x + threadIdx.x;
    if (i < n) p[i] = 0;
}

// ---------------------------------------------------------------------------
// patchify: x (B,C,512,512) -> A (S,B,768) bf16 ; k = p1*48 + p2*3 + c
// ---------------------------------------------------------------------------
__global__ void patchify_kernel(const float* __restrict__ x,
                                __bf16* __restrict__ A) {
    int idx = blockIdx.x * blockDim.x + threadIdx.x;   // < 16*3*512*512
    int b   = idx / 786432;            // 3*512*512
    int r   = idx % 786432;
    int c   = r / 262144;              // 512*512
    int pix = r % 262144;
    int iw  = pix >> 9;
    int ih  = pix & 511;
    int w = iw >> 4, p1 = iw & 15, h = ih >> 4, p2 = ih & 15;
    int s = w * 32 + h;
    int k = p1 * 48 + p2 * 3 + c;
    A[((size_t)(s * BATCH + b)) * DIN + k] = (__bf16)x[idx];
}

// ---------------------------------------------------------------------------
// WMMA fragment helpers (CDNA5 16x16x32 bf16, wave32)
// A (16x32): lane m=lane&15; elem e: k = (lane>=16?16:0)+e  (contiguous)
// B (32x16): lane n=lane&15; elem e: k = (lane>=16?16:0)+e  (contiguous)
// ---------------------------------------------------------------------------
__device__ __forceinline__ v16bf load_a_frag_g(const __bf16* __restrict__ A,
                                               int Mt, int kt, int lane) {
    int m   = lane & 15;
    int hi8 = (lane >> 4) * 8;
    const __bf16* row = A + (size_t)(Mt * 16 + m) * DIN + kt * 32;
    union { v16bf v; v8bf h[2]; } u;
    u.h[0] = *(const v8bf*)(row + 2 * hi8);
    u.h[1] = *(const v8bf*)(row + 2 * hi8 + 8);
    return u.v;
}

__device__ __forceinline__ v16bf load_b_frag_g(const __bf16* __restrict__ W,
                                               int ldk, int Nt, int kt, int lane) {
    int n   = lane & 15;
    int khi = (lane >> 4) * 16;
    return *(const v16bf*)(W + (size_t)(Nt * 16 + n) * ldk + kt * 32 + khi);
}

// ---------------------------------------------------------------------------
// xg GEMM: xg[d, s*16+b, g] = A[s*16+b, :] . Wih[d, g, :] + b_ih[d, g]
// M=16384, N=1152 (72 tiles), K=768 (24 tiles). 256 thr (8 waves).
// ---------------------------------------------------------------------------
__global__ void __launch_bounds__(256)
gemm_xg_kernel(const __bf16* __restrict__ A,
               const __bf16* __restrict__ Wih_layer,  // (2,1152,768) bf16
               const float*  __restrict__ bih_layer,  // (2,1152)
               float* __restrict__ xg) {               // (2,16384,1152)
    int d = blockIdx.z;
    const __bf16* W    = Wih_layer + (size_t)d * G3 * DIN;
    const float*  bias = bih_layer + d * G3;
    float*        xgo  = xg + (size_t)d * MROWS * G3;

    int lane = threadIdx.x & 31;
    int w    = threadIdx.x >> 5;
    int Mt   = blockIdx.x * 2 + (w >> 2);
    int Nt0  = blockIdx.y * 12 + (w & 3) * 3;

    v8f acc[3] = {};
    for (int kt = 0; kt < 24; ++kt) {
        v16bf a = load_a_frag_g(A, Mt, kt, lane);
        #pragma unroll
        for (int t = 0; t < 3; ++t) {
            v16bf b = load_b_frag_g(W, DIN, Nt0 + t, kt, lane);
            acc[t] = __builtin_amdgcn_wmma_f32_16x16x32_bf16(
                false, a, false, b, (short)0, acc[t], false, false);
        }
    }
    int n  = lane & 15;
    int m0 = (lane >= 16) ? 8 : 0;
    #pragma unroll
    for (int t = 0; t < 3; ++t) {
        int g = (Nt0 + t) * 16 + n;
        float bv = bias[g];
        #pragma unroll
        for (int v = 0; v < 8; ++v) {
            int row = Mt * 16 + v + m0;
            xgo[(size_t)row * G3 + g] = acc[t][v] + bv;
        }
    }
}

// ---------------------------------------------------------------------------
// GRU scan, slice-parallel: grid = (NSLICE, 2 dirs), 576 threads (18 waves).
// Slice ws owns j in [96*ws, 96*ws+96): W_hh rows {j, 384+j, 768+j} =
// 288x384 bf16 = 216KB, LDS-resident. Per step:
//   phase A: 18 N-tiles x 12 K WMMAs, B-frags from LDS (ds_load_b128)
//   phase B: gate math for own j-slice; h slice -> LDS frag image + global
//   4-block grid sync (arrival counter) + refresh full h frag image
// LDS: W 221184 | pre 16x288 f32 18432 | hS 16x96 f32 6144 | hA 12288
//      total 258048 B  (CDNA5: 320KB per workgroup)
// ---------------------------------------------------------------------------
__global__ void __launch_bounds__(SCAN_THREADS)
gru_scan_kernel(const float*  __restrict__ xg,        // (2,16384,1152)
                const __bf16* __restrict__ Whh_layer, // (2,1152,384) bf16
                const float*  __restrict__ bhh_layer, // (2,1152)
                __bf16* __restrict__ Anext,           // (S,B,768) bf16
                float*  __restrict__ out,             // (B,S,768) f32
                int final_layer,
                __bf16* __restrict__ g_hA,            // (2 parity, 2 dir, 6144) bf16
                int*    __restrict__ cnt) {           // (2 dir, 1024) arrival counters
    extern __shared__ char smem[];
    __bf16* Wl  = (__bf16*)smem;                 // 288*384 halves
    float*  pre = (float*)(smem + 221184);       // 16*288
    float*  hS  = (float*)(smem + 239616);       // 16*96
    __bf16* hA  = (__bf16*)(smem + 245760);      // 12*512 halves (full h image)

    const int ws  = blockIdx.x;                  // slice 0..3
    const int dir = blockIdx.y;                  // 0 fwd, 1 bwd
    const __bf16* W   = Whh_layer + (size_t)dir * G3 * HID;
    const float*  bh  = bhh_layer + dir * G3;
    const float*  xgd = xg + (size_t)dir * MROWS * G3;
    int* mycnt = cnt + dir * SEQ;

    const int tid  = threadIdx.x;
    const int lane = tid & 31;
    const int wv   = tid >> 5;                   // wave id 0..17

    // ---- stage W_hh slice into LDS (16B chunks): lr = gate*96 + jj ----
    for (int c = tid; c < 288 * 48; c += SCAN_THREADS) {
        int lr = c / 48, cw = c % 48;
        int g  = (lr / JSL) * HID + JSL * ws + (lr % JSL);
        *(float4*)(Wl + lr * HID + cw * 8) =
            *(const float4*)(W + (size_t)g * HID + cw * 8);
    }
    for (int i = tid; i < BATCH * JSL; i += SCAN_THREADS) hS[i] = 0.0f;
    for (int i = tid; i < 12 * 512;   i += SCAN_THREADS) hA[i] = (__bf16)0.0f;
    __syncthreads();

    // wave wv -> tile: gate = wv/6, o = wv%6 ; global Ntile = gate*24 + 6*ws + o
    const int gate = wv / 6, o = wv % 6;
    const int n    = lane & 15;
    const int m0   = (lane >= 16) ? 8 : 0;
    const int khi  = (lane >> 4) * 16;
    const int lr0  = gate * JSL + o * 16 + n;          // LDS weight row
    const int gG   = gate * HID + JSL * ws + o * 16 + n; // global gate row
    const float bv = bh[gG];

    for (int step = 0; step < SEQ; ++step) {
        int s = dir ? (SEQ - 1 - step) : step;

        // ---- phase A: pre-slice = h . Whh_sliceᵀ + b_hh  (WMMA, B from LDS)
        v8f acc = {};
        #pragma unroll
        for (int kt = 0; kt < 12; ++kt) {
            v16bf a = *(const v16bf*)(hA + kt * 512 + lane * 16);
            v16bf b = *(const v16bf*)(Wl + (size_t)lr0 * HID + kt * 32 + khi);
            acc = __builtin_amdgcn_wmma_f32_16x16x32_bf16(
                false, a, false, b, (short)0, acc, false, false);
        }
        #pragma unroll
        for (int v = 0; v < 8; ++v)
            pre[(v + m0) * 288 + gate * JSL + o * 16 + n] = acc[v] + bv;
        __syncthreads();

        // ---- phase B: gates + state update for own j-slice ----
        __bf16* gdst = g_hA + ((size_t)(step & 1) * 2 + dir) * 6144;
        #pragma unroll
        for (int i = 0; i < 3; ++i) {
            int idx = tid + i * SCAN_THREADS;        // 0..1727
            if (idx < BATCH * JSL) {
                int b  = idx / JSL;
                int jl = idx % JSL;
                int j  = JSL * ws + jl;
                const float* xv = xgd + (size_t)(s * BATCH + b) * G3;
                float xr = xv[j], xz = xv[HID + j], xn = xv[2 * HID + j];
                float hr = pre[b * 288 + jl];
                float hz = pre[b * 288 + JSL + jl];
                float hn = pre[b * 288 + 2 * JSL + jl];
                float rr = 1.0f / (1.0f + __expf(-(xr + hr)));
                float zz = 1.0f / (1.0f + __expf(-(xz + hz)));
                float nn = tanhf(xn + rr * hn);
                float hnew = (1.0f - zz) * nn + zz * hS[b * JSL + jl];
                hS[b * JSL + jl] = hnew;

                // fragment-image coords (kt in [3*ws, 3*ws+3) by construction)
                int kt = j >> 5, kl = j & 31;
                int lane2 = b + ((kl & 16) ? 16 : 0);
                int e     = kl & 15;
                int off   = kt * 512 + lane2 * 16 + e;
                __bf16 hb = (__bf16)hnew;
                hA[off]   = hb;
                gdst[off] = hb;

                Anext[((size_t)s * BATCH + b) * DIN + dir * HID + j] = hb;
                if (final_layer)
                    out[((size_t)b * SEQ + s) * DIN + dir * HID + j] = hnew;
            }
        }
        if (step < SEQ - 1) {
            int sn = dir ? (SEQ - 2 - step) : (step + 1);
            __builtin_prefetch(xgd + (size_t)(sn * BATCH + (tid & 15)) * G3
                                   + JSL * ws + (tid >> 4) * 6, 0, 0);
        }

        // ---- grid sync over the 4 slice-blocks of this direction ----
        __threadfence();
        __syncthreads();
        if (tid == 0) {
            __hip_atomic_fetch_add(&mycnt[step], 1,
                                   __ATOMIC_RELEASE, __HIP_MEMORY_SCOPE_AGENT);
            while (__hip_atomic_load(&mycnt[step],
                                     __ATOMIC_ACQUIRE, __HIP_MEMORY_SCOPE_AGENT)
                   < NSLICE)
                __builtin_amdgcn_s_sleep(1);
        }
        __syncthreads();
        __threadfence();

        // ---- refresh full h fragment image from global exchange buffer ----
        const unsigned* gsrc = (const unsigned*)gdst;
        unsigned* hAd = (unsigned*)hA;
        for (int i = tid; i < 3072; i += SCAN_THREADS) hAd[i] = gsrc[i];
        __syncthreads();
    }
}

// ---------------------------------------------------------------------------
// host launcher
// ---------------------------------------------------------------------------
extern "C" void kernel_launch(void* const* d_in, const int* in_sizes, int n_in,
                              void* d_out, int out_size, void* d_ws, size_t ws_size,
                              hipStream_t stream) {
    const float* x    = (const float*)d_in[0];   // (16,3,512,512)
    const float* W_ih = (const float*)d_in[1];   // (3,2,1152,768)
    const float* W_hh = (const float*)d_in[2];   // (3,2,1152,384)
    const float* b_ih = (const float*)d_in[3];   // (3,2,1152)
    const float* b_hh = (const float*)d_in[4];   // (3,2,1152)
    float* out = (float*)d_out;                  // (16,1024,768)

    char* ws = (char*)d_ws;
    size_t off = 0;
    __bf16* A0    = (__bf16*)(ws + off); off += (size_t)MROWS * DIN * 2;
    __bf16* A1    = (__bf16*)(ws + off); off += (size_t)MROWS * DIN * 2;
    __bf16* Wih_b = (__bf16*)(ws + off); off += (size_t)6 * G3 * DIN * 2;
    __bf16* Whh_b = (__bf16*)(ws + off); off += (size_t)6 * G3 * HID * 2;
    float*  xg    = (float*)(ws + off);  off += (size_t)2 * MROWS * G3 * 4;
    __bf16* g_hA  = (__bf16*)(ws + off); off += (size_t)2 * 2 * 6144 * 2;   // 49KB
    int*    cnt   = (int*)(ws + off);    off += (size_t)3 * 2 * SEQ * 4;    // 24KB

    const int nWih = 6 * G3 * DIN;   // 5,308,416
    const int nWhh = 6 * G3 * HID;   // 2,654,208
    cvt_f32_bf16_kernel<<<(nWih + 255) / 256, 256, 0, stream>>>(W_ih, Wih_b, nWih);
    cvt_f32_bf16_kernel<<<(nWhh + 255) / 256, 256, 0, stream>>>(W_hh, Whh_b, nWhh);
    zero_i32_kernel<<<(3 * 2 * SEQ + 255) / 256, 256, 0, stream>>>(cnt, 3 * 2 * SEQ);

    patchify_kernel<<<(16 * 3 * 512 * 512) / 256, 256, 0, stream>>>(x, A0);

    __bf16* bufs[2] = {A0, A1};
    for (int l = 0; l < 3; ++l) {
        __bf16* Ain  = bufs[l & 1];
        __bf16* Aout = bufs[(l + 1) & 1];
        gemm_xg_kernel<<<dim3(MROWS / 32, 6, 2), 256, 0, stream>>>(
            Ain, Wih_b + (size_t)l * 2 * G3 * DIN, b_ih + l * 2 * G3, xg);
        gru_scan_kernel<<<dim3(NSLICE, 2), SCAN_THREADS, 258048, stream>>>(
            xg, Whh_b + (size_t)l * 2 * G3 * HID, b_hh + l * 2 * G3,
            Aout, out, (l == 2) ? 1 : 0, g_hA, cnt + l * 2 * SEQ);
    }
}
